// SpecDetrTransformerEncoder_87883620811064
// MI455X (gfx1250) — matrix operand order repaired and checked by reading, not compiled, not run on hardware
//
#include <hip/hip_runtime.h>
#include <stdint.h>

// ---------------- problem constants (from the reference) ----------------
#define NLAYERS 6
#define E_DIM   256
#define NHEAD   8
#define HDIM    32
#define NLVL    4
#define NPTS    4
#define FF_DIM  1024
#define BS_N    2
#define NQ_N    17821              // 100*134 + 50*67 + 25*34 + 13*17
#define M_TOT   (BS_N * NQ_N)      // 35642 rows for all GEMMs

typedef __attribute__((ext_vector_type(16))) __bf16 v16bf;
typedef __attribute__((ext_vector_type(8)))  float  v8f;
typedef unsigned short u16;

__constant__ int cH[4]   = {100, 50, 25, 13};
__constant__ int cW[4]   = {134, 67, 34, 17};
__constant__ int cLSI[4] = {0, 13400, 16750, 17600};

__device__ __forceinline__ u16 f2bf(float f) {
    union { float f; uint32_t u; } x; x.f = f;
    uint32_t r = x.u + 0x7fffu + ((x.u >> 16) & 1u);   // round-to-nearest-even
    return (u16)(r >> 16);
}

union FragB {            // one WMMA 16-bit operand: 8 VGPRs / 16 bf16 per lane
    uint4    q[2];
    uint32_t u[8];
    v16bf    v;
};

// low 32 bits of a generic shared-memory address == LDS byte offset
__device__ __forceinline__ uint32_t lds_off(const void* p) {
    return (uint32_t)(uintptr_t)p;
}
// CDNA5 async global->LDS copy, 16B per lane, tracked by ASYNCcnt
__device__ __forceinline__ void async_ld_b128(uint32_t dst_lds, const void* gaddr) {
    asm volatile("global_load_async_to_lds_b128 %0, %1, off"
                 :: "v"(dst_lds), "v"(gaddr) : "memory");
}
__device__ __forceinline__ void wait_async_le4() {
    asm volatile("s_wait_asynccnt 0x4" ::: "memory");
}
__device__ __forceinline__ void wait_async_le2() {
    asm volatile("s_wait_asynccnt 0x2" ::: "memory");
}
__device__ __forceinline__ void wait_async_le0() {
    asm volatile("s_wait_asynccnt 0x0" ::: "memory");
}

// ---------------------------------------------------------------------------
// WMMA bf16 GEMM:  C[M,N] = A[M,K](bf16) * B^T  with B pre-transposed to
// [N,K] bf16 so both LDS tiles are straight row-major copies (async b128).
// Block tile 128x128x32, 8 waves, each wave 32x64 (2x4 wmma fragments).
// Double-buffered LDS: tile i+1 is async-loading while tile i is computed;
// s_wait_asynccnt <= 4 (in-order async completion) pins the landed tile.
// ---------------------------------------------------------------------------
#define BM 128
#define BN 128
#define BK 32

__global__ __launch_bounds__(256)
void gemm_bf16_kernel(const u16* __restrict__ A, const u16* __restrict__ BwT,
                      const float* __restrict__ bias,
                      float* __restrict__ Cf, u16* __restrict__ Cbf,
                      const unsigned char* __restrict__ rowMask,
                      int M, int N, int K, int relu, int outBf)
{
    __shared__ uint32_t lds_a[2][BM * BK / 2];   // 2 x 8KB
    __shared__ uint32_t lds_b[2][BN * BK / 2];   // 2 x 8KB

    const int tid  = threadIdx.x;
    const int wave = tid >> 5;
    const int lane = tid & 31;
    const int lrow = lane & 15;
    const int half = lane >> 4;

    const int wm = wave & 3;     // 4 row-blocks of 32
    const int wn = wave >> 2;    // 2 col-blocks of 64

    const int m_blk = blockIdx.y * BM;
    const int n_blk = blockIdx.x * BN;
    const bool fullM = (m_blk + BM) <= M;

    // each thread copies 2 x 16B chunks of each tile: chunk c -> row c>>2, kq c&3
    const int r0 = tid >> 2, kq0 = tid & 3;           // chunk tid
    const int r1 = r0 + 64;                           // chunk tid+256

    auto fill_tile = [&](int kt, int buf) {
        if (fullM) {
            async_ld_b128(lds_off(&lds_a[buf][(r0 << 4) + kq0 * 4]),
                          A + (size_t)(m_blk + r0) * K + kt + kq0 * 8);
            async_ld_b128(lds_off(&lds_a[buf][(r1 << 4) + kq0 * 4]),
                          A + (size_t)(m_blk + r1) * K + kt + kq0 * 8);
        } else {
            #pragma unroll
            for (int i = tid; i < BM * BK / 2; i += 256) {
                int row = i >> 4, kp = i & 15;
                int gr = m_blk + row;
                uint32_t val = 0u;
                if (gr < M)
                    val = *(const uint32_t*)(A + (size_t)gr * K + kt + kp * 2);
                lds_a[buf][(row << 4) + kp] = val;
            }
        }
        async_ld_b128(lds_off(&lds_b[buf][(r0 << 4) + kq0 * 4]),
                      BwT + (size_t)(n_blk + r0) * K + kt + kq0 * 8);
        async_ld_b128(lds_off(&lds_b[buf][(r1 << 4) + kq0 * 4]),
                      BwT + (size_t)(n_blk + r1) * K + kt + kq0 * 8);
    };

    v8f zero = {0.f,0.f,0.f,0.f,0.f,0.f,0.f,0.f};
    v8f acc[2][4];
    #pragma unroll
    for (int r = 0; r < 2; ++r)
        #pragma unroll
        for (int c = 0; c < 4; ++c) acc[r][c] = zero;

    const int nk = K / BK;
    fill_tile(0, 0);                                  // prologue: tile 0 in flight

    for (int i = 0; i < nk; ++i) {
        const int cur = i & 1;
        if (i + 1 < nk) {
            fill_tile((i + 1) * BK, cur ^ 1);         // next tile in flight
            // L2 prefetch two tiles ahead
            if (i + 2 < nk) {
                int kpf = (i + 2) * BK;
                int pr = m_blk + (tid >> 1);
                if (pr < M)
                    __builtin_prefetch(A + (size_t)pr * K + kpf + (tid & 1) * 16, 0, 3);
                __builtin_prefetch(BwT + (size_t)(n_blk + (tid >> 1)) * K + kpf + (tid & 1) * 16, 0, 3);
            }
            if (fullM) wait_async_le4();              // tile i landed; i+1 may fly
            else       wait_async_le2();
        } else {
            wait_async_le0();                         // final tile landed
        }
        __syncthreads();                              // tile i visible to all waves

        // ---- load fragments per ISA lane layouts ----
        FragB af[2], bfr[4];
        #pragma unroll
        for (int r = 0; r < 2; ++r) {
            int row = wm * 32 + r * 16 + lrow;        // A: lane = M row
            int ab  = (row << 4) + (half << 2);       // K = half*8..+7, then +16
            af[r].q[0] = *(const uint4*)&lds_a[cur][ab];
            af[r].q[1] = *(const uint4*)&lds_a[cur][ab + 8];
        }
        #pragma unroll
        for (int c = 0; c < 4; ++c) {
            int n  = wn * 64 + c * 16 + lrow;         // B: lane = N col
            int bb = (n << 4) + (half << 2);          // K = half*8..+7, then +16
            bfr[c].q[0] = *(const uint4*)&lds_b[cur][bb];
            bfr[c].q[1] = *(const uint4*)&lds_b[cur][bb + 8];
        }
        #pragma unroll
        for (int r = 0; r < 2; ++r)
            #pragma unroll
            for (int c = 0; c < 4; ++c)
                acc[r][c] = __builtin_amdgcn_wmma_f32_16x16x32_bf16(
                    false, af[r].v, false, bfr[c].v, (short)0, acc[r][c], false, false);

        __syncthreads();                              // reads done before buf reuse
    }

    // ---- epilogue: bias (+relu) (+mask) (+bf16 cast) ----
    #pragma unroll
    for (int r = 0; r < 2; ++r) {
        #pragma unroll
        for (int c = 0; c < 4; ++c) {
            int col = n_blk + wn * 64 + c * 16 + lrow;
            float bv = bias ? bias[col] : 0.f;
            #pragma unroll
            for (int i = 0; i < 8; ++i) {
                int row = m_blk + wm * 32 + r * 16 + half * 8 + i;
                if (row < M) {
                    float v = acc[r][c][i] + bv;
                    if (relu) v = fmaxf(v, 0.f);
                    if (rowMask && rowMask[row]) v = 0.f;
                    if (outBf) Cbf[(size_t)row * N + col] = f2bf(v);
                    else       Cf [(size_t)row * N + col] = v;
                }
            }
        }
    }
}

// ---------------------------------------------------------------------------
// weight convert + transpose:  src [L,K,N] f32  ->  dst [L,N,K] bf16
// ---------------------------------------------------------------------------
__global__ __launch_bounds__(256)
void convT_bf16_kernel(const float* __restrict__ src, u16* __restrict__ dst,
                       int K, int N, long total)
{
    long i = (long)blockIdx.x * 256 + threadIdx.x;
    if (i >= total) return;
    long kn = (long)K * N;
    long l  = i / kn;
    long r  = i - l * kn;
    int  k  = (int)(r / N);
    int  n  = (int)(r - (long)k * N);
    dst[l * kn + (long)n * K + k] = f2bf(src[i]);
}

__global__ __launch_bounds__(256)
void prep_kernel(const float* __restrict__ q, const float* __restrict__ qpos,
                 u16* __restrict__ qbf, u16* __restrict__ qpbf, long n)
{
    long i = (long)blockIdx.x * 256 + threadIdx.x;
    if (i < n) {
        float a = q[i];
        qbf[i]  = f2bf(a);
        qpbf[i] = f2bf(a + qpos[i]);
    }
}

// reference points: ref[b,q,lvl,{x,y}] = base(own level) * valid_ratios[b,lvl]
__global__ __launch_bounds__(256)
void refpoints_kernel(const float* __restrict__ vr, float* __restrict__ refp)
{
    long i = (long)blockIdx.x * 256 + threadIdx.x;    // over BS*NQ
    if (i >= (long)BS_N * NQ_N) return;
    int b = (int)(i / NQ_N), q = (int)(i % NQ_N);
    int L = (q < 13400) ? 0 : (q < 16750) ? 1 : (q < 17600) ? 2 : 3;
    int pos = q - cLSI[L];
    int y = pos / cW[L], x = pos % cW[L];
    float rxb = (x + 0.5f) / (vr[(b * NLVL + L) * 2 + 0] * (float)cW[L]);
    float ryb = (y + 0.5f) / (vr[(b * NLVL + L) * 2 + 1] * (float)cH[L]);
    #pragma unroll
    for (int l2 = 0; l2 < NLVL; ++l2) {
        refp[(i * NLVL + l2) * 2 + 0] = rxb * vr[(b * NLVL + l2) * 2 + 0];
        refp[(i * NLVL + l2) * 2 + 1] = ryb * vr[(b * NLVL + l2) * 2 + 1];
    }
}

// ---------------------------------------------------------------------------
// deformable sampling: one wave per (b,q,h); lane == head dim (HD==wave32)
// softmax(16 logits) fused; bilinear gather from v[b,nq,h,hd]; bf16 output
// (feeds the out-projection WMMA directly).
// ---------------------------------------------------------------------------
__global__ __launch_bounds__(256)
void sampling_kernel(const float* __restrict__ v, const float* __restrict__ off,
                     const float* __restrict__ aw, const float* __restrict__ refp,
                     u16* __restrict__ out_bf)
{
    const int warp = threadIdx.x >> 5;
    const int lane = threadIdx.x & 31;
    long g = (long)blockIdx.x * 8 + warp;
    const long total = (long)BS_N * NQ_N * NHEAD;
    if (g >= total) return;
    int  h  = (int)(g % NHEAD);
    long bq = g / NHEAD;                        // b*NQ + q
    long b  = bq / NQ_N;

    const float* awp = aw + bq * (NHEAD * NLVL * NPTS) + h * 16;
    float wgt16[16]; float mx = -1e30f;
    #pragma unroll
    for (int j = 0; j < 16; ++j) { wgt16[j] = awp[j]; mx = fmaxf(mx, wgt16[j]); }
    float s = 0.f;
    #pragma unroll
    for (int j = 0; j < 16; ++j) { wgt16[j] = __expf(wgt16[j] - mx); s += wgt16[j]; }
    float inv = 1.f / s;

    const float* offp = off + bq * 256 + h * 32;   // [lvl][p][2]
    const float* rp   = refp + bq * (NLVL * 2);
    float accv = 0.f;

    #pragma unroll
    for (int lvl = 0; lvl < NLVL; ++lvl) {
        const int H = cH[lvl], W = cW[lvl], S = cLSI[lvl];
        const float rx = rp[lvl * 2 + 0], ry = rp[lvl * 2 + 1];
        const float* vl = v + (((b * NQ_N + S) * NHEAD + h) * HDIM) + lane;
        #pragma unroll
        for (int p = 0; p < NPTS; ++p) {
            float ox = offp[(lvl * NPTS + p) * 2 + 0];
            float oy = offp[(lvl * NPTS + p) * 2 + 1];
            float lx = rx + ox / (float)W;
            float ly = ry + oy / (float)H;
            float x = lx * W - 0.5f, y = ly * H - 0.5f;
            float x0 = floorf(x), y0 = floorf(y);
            float samp = 0.f;
            #pragma unroll
            for (int dx = 0; dx < 2; ++dx)
                #pragma unroll
                for (int dy = 0; dy < 2; ++dy) {
                    float xi = x0 + dx, yi = y0 + dy;
                    if (xi >= 0.f && xi < (float)W && yi >= 0.f && yi < (float)H) {
                        float wb = (1.f - fabsf(x - xi)) * (1.f - fabsf(y - yi));
                        long idx = (long)((int)yi * W + (int)xi);
                        samp += wb * vl[idx * (NHEAD * HDIM)];
                    }
                }
            accv += (wgt16[lvl * NPTS + p] * inv) * samp;
        }
    }
    out_bf[bq * E_DIM + h * HDIM + lane] = f2bf(accv);
}

// residual add + layernorm over E=256, one wave per row; optional bf16 copy
__global__ __launch_bounds__(256)
void addln_kernel(float* __restrict__ q, const float* __restrict__ add,
                  const float* __restrict__ ga, const float* __restrict__ be,
                  u16* __restrict__ qbf_out, int M)
{
    const int warp = threadIdx.x >> 5, lane = threadIdx.x & 31;
    long row = (long)blockIdx.x * 8 + warp;
    if (row >= M) return;
    const float* qr = q   + row * E_DIM;
    const float* ar = add + row * E_DIM;
    float x[8]; float s = 0.f;
    #pragma unroll
    for (int j = 0; j < 8; ++j) { x[j] = qr[j * 32 + lane] + ar[j * 32 + lane]; s += x[j]; }
    #pragma unroll
    for (int o = 16; o > 0; o >>= 1) s += __shfl_xor(s, o, 32);
    float m = s * (1.f / 256.f);
    float vv = 0.f;
    #pragma unroll
    for (int j = 0; j < 8; ++j) { float d = x[j] - m; vv += d * d; }
    #pragma unroll
    for (int o = 16; o > 0; o >>= 1) vv += __shfl_xor(vv, o, 32);
    float r = rsqrtf(vv * (1.f / 256.f) + 1e-5f);
    #pragma unroll
    for (int j = 0; j < 8; ++j) {
        int col = j * 32 + lane;
        float o2 = (x[j] - m) * r * ga[col] + be[col];
        q[row * E_DIM + col] = o2;
        if (qbf_out) qbf_out[row * E_DIM + col] = f2bf(o2);
    }
}

// ---------------------------------------------------------------------------
extern "C" void kernel_launch(void* const* d_in, const int* in_sizes, int n_in,
                              void* d_out, int out_size, void* d_ws, size_t ws_size,
                              hipStream_t stream)
{
    (void)in_sizes; (void)n_in; (void)out_size; (void)ws_size;
    const float* query  = (const float*)d_in[0];
    const float* qpos   = (const float*)d_in[1];
    const unsigned char* kpm = (const unsigned char*)d_in[2];
    const float* vr     = (const float*)d_in[5];
    const float* w_off  = (const float*)d_in[6];
    const float* b_off  = (const float*)d_in[7];
    const float* w_attn = (const float*)d_in[8];
    const float* b_attn = (const float*)d_in[9];
    const float* w_val  = (const float*)d_in[10];
    const float* b_val  = (const float*)d_in[11];
    const float* w_out  = (const float*)d_in[12];
    const float* b_out  = (const float*)d_in[13];
    const float* w_ffn1 = (const float*)d_in[14];
    const float* b_ffn1 = (const float*)d_in[15];
    const float* w_ffn2 = (const float*)d_in[16];
    const float* b_ffn2 = (const float*)d_in[17];
    const float* ln1_g  = (const float*)d_in[18];
    const float* ln1_b  = (const float*)d_in[19];
    const float* ln2_g  = (const float*)d_in[20];
    const float* ln2_b  = (const float*)d_in[21];

    float* q = (float*)d_out;                 // running activation buffer

    // workspace carve-up
    char* ws = (char*)d_ws; size_t o = 0;
    auto carve = [&](size_t bytes) -> char* {
        char* p = ws + o; o = (o + bytes + 255) & ~(size_t)255; return p;
    };
    u16*   wvT   = (u16*)  carve((size_t)NLAYERS * E_DIM * E_DIM  * 2);  // [N,K]
    u16*   woT   = (u16*)  carve((size_t)NLAYERS * E_DIM * E_DIM  * 2);
    u16*   woffT = (u16*)  carve((size_t)NLAYERS * E_DIM * 256    * 2);
    u16*   watT  = (u16*)  carve((size_t)NLAYERS * E_DIM * 128    * 2);
    u16*   wf1T  = (u16*)  carve((size_t)NLAYERS * E_DIM * FF_DIM * 2);
    u16*   wf2T  = (u16*)  carve((size_t)NLAYERS * FF_DIM * E_DIM * 2);
    u16*   qbf   = (u16*)  carve((size_t)M_TOT * E_DIM * 2);
    u16*   qpbf  = (u16*)  carve((size_t)M_TOT * E_DIM * 2);   // reused for bf16 samples
    u16*   hid_bf= (u16*)  carve((size_t)M_TOT * FF_DIM * 2);
    float* gout  = (float*)carve((size_t)M_TOT * E_DIM * 4);   // v / proj / ffn2 outputs
    float* offb  = (float*)carve((size_t)M_TOT * 256 * 4);
    float* awb   = (float*)carve((size_t)M_TOT * 128 * 4);
    float* refb  = (float*)carve((size_t)BS_N * NQ_N * NLVL * 2 * 4);

    const long nQE = (long)M_TOT * E_DIM;
    auto convT = [&](const float* s, u16* d, int K, int N) {
        long tot = (long)NLAYERS * K * N;
        convT_bf16_kernel<<<dim3((unsigned)((tot + 255) / 256)), 256, 0, stream>>>(s, d, K, N, tot);
    };
    auto gemm = [&](const u16* A, const u16* BT, const float* bias, float* Cf, u16* Cbf,
                    const unsigned char* mask, int N, int K, int relu) {
        dim3 g((unsigned)(N / 128), (unsigned)((M_TOT + 127) / 128));
        gemm_bf16_kernel<<<g, 256, 0, stream>>>(A, BT, bias, Cf, Cbf, mask,
                                                M_TOT, N, K, relu, Cbf != nullptr);
    };

    // init: q <- query; weights -> bf16 transposed; reference points
    hipMemcpyAsync(q, query, nQE * sizeof(float), hipMemcpyDeviceToDevice, stream);
    convT(w_val , wvT  , E_DIM, E_DIM);
    convT(w_out , woT  , E_DIM, E_DIM);
    convT(w_off , woffT, E_DIM, 256);
    convT(w_attn, watT , E_DIM, 128);
    convT(w_ffn1, wf1T , E_DIM, FF_DIM);
    convT(w_ffn2, wf2T , FF_DIM, E_DIM);
    refpoints_kernel<<<dim3((unsigned)(((long)BS_N * NQ_N + 255) / 256)), 256, 0, stream>>>(vr, refb);

    const unsigned sampBlocks = (unsigned)(((long)BS_N * NQ_N * NHEAD + 7) / 8);
    const unsigned lnBlocks   = (unsigned)((M_TOT + 7) / 8);
    const unsigned elBlocks   = (unsigned)((nQE + 255) / 256);

    for (int l = 0; l < NLAYERS; ++l) {
        // q, q+pos -> bf16
        prep_kernel<<<elBlocks, 256, 0, stream>>>(q, qpos, qbf, qpbf, nQE);
        // v = q @ w_val + b_val  (padding mask fused)
        gemm(qbf, wvT + (size_t)l * E_DIM * E_DIM, b_val + l * E_DIM,
             gout, nullptr, kpm, E_DIM, E_DIM, 0);
        // offsets / attention logits from q+pos
        gemm(qpbf, woffT + (size_t)l * E_DIM * 256, b_off + l * 256,
             offb, nullptr, nullptr, 256, E_DIM, 0);
        gemm(qpbf, watT + (size_t)l * E_DIM * 128, b_attn + l * 128,
             awb, nullptr, nullptr, 128, E_DIM, 0);
        // deformable sampling + softmax -> bf16 (qpbf reused as A of out-proj)
        sampling_kernel<<<sampBlocks, 256, 0, stream>>>(gout, offb, awb, refb, qpbf);
        // out projection
        gemm(qpbf, woT + (size_t)l * E_DIM * E_DIM, b_out + l * E_DIM,
             gout, nullptr, nullptr, E_DIM, E_DIM, 0);
        addln_kernel<<<lnBlocks, 256, 0, stream>>>(q, gout, ln1_g + l * E_DIM, ln1_b + l * E_DIM,
                                                   qbf, M_TOT);   // qbf = bf16(post-LN1)
        // FFN (hidden kept bf16 straight from GEMM epilogue)
        gemm(qbf, wf1T + (size_t)l * E_DIM * FF_DIM, b_ffn1 + l * FF_DIM,
             nullptr, hid_bf, nullptr, FF_DIM, E_DIM, 1);
        gemm(hid_bf, wf2T + (size_t)l * FF_DIM * E_DIM, b_ffn2 + l * E_DIM,
             gout, nullptr, nullptr, E_DIM, FF_DIM, 0);
        addln_kernel<<<lnBlocks, 256, 0, stream>>>(q, gout, ln2_g + l * E_DIM, ln2_b + l * E_DIM,
                                                   nullptr, M_TOT);
    }
}
